// BlockCirculantConv_7799660609611
// MI455X (gfx1250) — compile-verified, matrix operand order checked by reading, not run on hardware
//
#include <hip/hip_runtime.h>
#include <hip/hip_bf16.h>
#include <stdint.h>

// ---- problem constants ----
#define MROWS   8192          // 8 batches * 1024 positions
#define NCOLS   512           // OUT_FEATURES
#define KDIM    2304          // 9*256
#define BM      128
#define BN      64
#define BK      32
#define LDSPAD  40            // halfs per LDS row (80B stride, multiple of 16B)

typedef _Float16 half8  __attribute__((ext_vector_type(8)));
typedef _Float16 half16 __attribute__((ext_vector_type(16)));
typedef float    floatv8 __attribute__((ext_vector_type(8)));

// ---------------- async global->LDS copy (CDNA5 ASYNCcnt path) ----------------
__device__ __forceinline__ void cp_async16(const _Float16* g, _Float16* l) {
  unsigned lds_off = (unsigned)(uintptr_t)l;             // low 32 bits of generic shared ptr = LDS offset
  unsigned long long ga = (unsigned long long)(uintptr_t)g;
  asm volatile("global_load_async_to_lds_b128 %0, %1, off"
               :
               : "v"(lds_off), "v"(ga)
               : "memory");
}

__device__ __forceinline__ void wait_async0() {
#if __has_builtin(__builtin_amdgcn_s_wait_asynccnt)
  __builtin_amdgcn_s_wait_asynccnt(0);
#else
  asm volatile("s_wait_asynccnt 0" ::: "memory");
#endif
}

// ---------------- kernel 1: im2col (with reference's flat reshape) -> f16 ----------------
__global__ __launch_bounds__(256) void bcc_im2col_f16(const float* __restrict__ x,
                                                      _Float16* __restrict__ X) {
  size_t idx = (size_t)blockIdx.x * blockDim.x + threadIdx.x;
  const size_t total = (size_t)MROWS * KDIM;
  if (idx >= total) return;
  int n  = (int)(idx / KDIM);
  int i  = (int)(idx - (size_t)n * KDIM);
  int b  = n >> 10;            // batch
  int nl = n & 1023;           // position chunk within batch
  int flat = nl * KDIM + i;    // flat index into (F=2304, L=1024) buffer
  int f = flat >> 10;          // feature row (c*9 + di*3 + dj)
  int l = flat & 1023;         // spatial position ho*32+wo
  int c  = f / 9;
  int tt = f - c * 9;
  int di = tt / 3;
  int dj = tt - di * 3;
  int ho = l >> 5, wo = l & 31;
  int hi = ho + di - 1;
  int wi = wo + dj - 1;
  float v = 0.0f;
  if ((unsigned)hi < 32u && (unsigned)wi < 32u)
    v = x[(((size_t)b * 256 + c) * 32 + hi) * 32 + wi];
  X[idx] = (_Float16)v;
}

// ---------------- kernel 2: circulant weight expansion -> f16 dense [512][2304] ----------------
__global__ __launch_bounds__(256) void bcc_wexpand_f16(const float* __restrict__ w,
                                                       _Float16* __restrict__ Wd) {
  int idx = blockIdx.x * blockDim.x + threadIdx.x;
  if (idx >= NCOLS * KDIM) return;
  int o = idx / KDIM;
  int k = idx - o * KDIM;
  int p = o >> 6, j = o & 63;
  int q = k >> 6, i = k & 63;
  int s = (j - i) & 63;
  Wd[idx] = (_Float16)w[(((p * 36) + q) << 6) + s];
}

// ---------------- kernel 3: WMMA f16 GEMM with async double-buffered LDS ----------------
// out[b, o, l] = sum_k X[b*1024 + l', k] * Wd[o, k]   (row m of X maps to (b=m>>10, l=m&1023))
__global__ __launch_bounds__(256) void bcc_wmma_gemm(const _Float16* __restrict__ X,
                                                     const _Float16* __restrict__ Wd,
                                                     float* __restrict__ out) {
  __shared__ __align__(16) _Float16 As[2][BM][LDSPAD];
  __shared__ __align__(16) _Float16 Bs[2][BN][LDSPAD];

  const int t    = threadIdx.x;
  const int lane = t & 31;
  const int wave = t >> 5;           // 0..7
  const int wm   = wave >> 1;        // 0..3  (M sub-tile of 32)
  const int wn   = wave & 1;         // 0..1  (N sub-tile of 32)
  const int rowBase = blockIdx.x * BM;
  const int colBase = blockIdx.y * BN;
  const int h   = lane >> 4;         // lane half (A/B fragment K-half select)
  const int m16 = lane & 15;

  floatv8 acc[2][2] = {};

  // stage one BK-slice: A = 128x32 halfs (512 x 16B chunks), B = 64x32 halfs (256 chunks)
  auto issue = [&](int buf, int kt) {
    const int k0 = kt * BK;
    int c   = t;                     // A chunk 0
    int row = c >> 2, sub = c & 3;
    cp_async16(X + (size_t)(rowBase + row) * KDIM + k0 + sub * 8, &As[buf][row][sub * 8]);
    c   = t + 256;                   // A chunk 1
    row = c >> 2; sub = c & 3;
    cp_async16(X + (size_t)(rowBase + row) * KDIM + k0 + sub * 8, &As[buf][row][sub * 8]);
    row = t >> 2; sub = t & 3;       // B chunk
    cp_async16(Wd + (size_t)(colBase + row) * KDIM + k0 + sub * 8, &Bs[buf][row][sub * 8]);
  };

  issue(0, 0);

  const int NK = KDIM / BK;          // 72
  for (int kt = 0; kt < NK; ++kt) {
    const int cur = kt & 1;
    wait_async0();                   // my slice-kt copies landed in LDS
    __syncthreads();                 // everyone's landed; everyone done reading buf^1
    if (kt + 1 < NK) issue(cur ^ 1, kt + 1);

    half16 afrag[2], bfrag[2];
#pragma unroll
    for (int tm = 0; tm < 2; ++tm) {
      int row = wm * 32 + tm * 16 + m16;
      half8 lo = *(const half8*)&As[cur][row][h * 8];
      half8 hi = *(const half8*)&As[cur][row][16 + h * 8];
      afrag[tm] = __builtin_shufflevector(lo, hi, 0,1,2,3,4,5,6,7,8,9,10,11,12,13,14,15);
    }
#pragma unroll
    for (int tn = 0; tn < 2; ++tn) {
      int col = wn * 32 + tn * 16 + m16;
      half8 lo = *(const half8*)&Bs[cur][col][h * 8];
      half8 hi = *(const half8*)&Bs[cur][col][16 + h * 8];
      bfrag[tn] = __builtin_shufflevector(lo, hi, 0,1,2,3,4,5,6,7,8,9,10,11,12,13,14,15);
    }
#pragma unroll
    for (int tm = 0; tm < 2; ++tm)
#pragma unroll
      for (int tn = 0; tn < 2; ++tn)
        acc[tm][tn] = __builtin_amdgcn_wmma_f32_16x16x32_f16(
            false, afrag[tm], false, bfrag[tn], (short)0, acc[tm][tn], false, false);
  }

  // epilogue: D layout — VGPR r holds M=r (lanes 0-15) / M=r+8 (lanes 16-31), N = lane&15.
  // scatter into (B, 512, 32, 32): out[(m>>10)*512*1024 + o*1024 + (m&1023)]
#pragma unroll
  for (int tm = 0; tm < 2; ++tm)
#pragma unroll
    for (int tn = 0; tn < 2; ++tn) {
      int o = colBase + wn * 32 + tn * 16 + m16;
#pragma unroll
      for (int r = 0; r < 8; ++r) {
        int m = rowBase + wm * 32 + tm * 16 + h * 8 + r;
        int b = m >> 10, l = m & 1023;
        out[((size_t)b * NCOLS + o) * 1024 + l] = acc[tm][tn][r];
      }
    }
}

extern "C" void kernel_launch(void* const* d_in, const int* in_sizes, int n_in,
                              void* d_out, int out_size, void* d_ws, size_t ws_size,
                              hipStream_t stream) {
  const float* x = (const float*)d_in[0];     // (8,256,32,32) f32
  const float* w = (const float*)d_in[1];     // (8,36,64) f32
  float* out = (float*)d_out;                 // (8,512,32,32) f32

  _Float16* Xbf = (_Float16*)d_ws;                                   // 8192*2304 f16 = 37.75 MB
  _Float16* Wd  = (_Float16*)((char*)d_ws + (size_t)MROWS * KDIM * sizeof(_Float16));

  {
    size_t total = (size_t)MROWS * KDIM;
    int blocks = (int)((total + 255) / 256);
    bcc_im2col_f16<<<blocks, 256, 0, stream>>>(x, Xbf);
  }
  {
    int total = NCOLS * KDIM;
    bcc_wexpand_f16<<<(total + 255) / 256, 256, 0, stream>>>(w, Wd);
  }
  {
    dim3 grid(MROWS / BM, NCOLS / BN);  // 64 x 8 = 512 workgroups
    bcc_wmma_gemm<<<grid, 256, 0, stream>>>(Xbf, Wd, out);
  }
  (void)in_sizes; (void)n_in; (void)out_size; (void)ws_size;
}
